// RecurrentProcessorCell_44753559225036
// MI455X (gfx1250) — matrix operand order repaired and checked by reading, not compiled, not run on hardware
//
#include <hip/hip_runtime.h>
#include <hip/hip_bf16.h>

// ---------------------------------------------------------------------------
// CDNA5 (gfx1250) GNN processor: bf16 WMMA core, fp32 epilogue.
// 64-row tiles: 4 independent WMMAs share each B fragment, halving L2
// weight traffic vs 32-row tiles. LDS ~96KB/block (CDNA5 WGP has 320KB).
// ---------------------------------------------------------------------------

typedef __attribute__((ext_vector_type(16))) __bf16 v16bf;
typedef __attribute__((ext_vector_type(8)))  __bf16 v8bf;
typedef __attribute__((ext_vector_type(4)))  __bf16 v4bf;
typedef __attribute__((ext_vector_type(8)))  float  v8f;

#define GNN_L 4
#define GNN_N 50000
#define GNN_E 250000
#define GNN_D 128

// D = A(16x32 bf16) * B(32x16 bf16) + C(16x16 f32)
__device__ __forceinline__ v8f wmma_bf16(v16bf a, v16bf b, v8f c) {
  return __builtin_amdgcn_wmma_f32_16x16x32_bf16(
      /*neg_a=*/false, a, /*neg_b=*/false, b,
      /*c_mod=*/(short)0, c, /*reuse_a=*/false, /*reuse_b=*/false);
}

// A fragment: 16x32 bf16 tile from a row-major bf16 buffer (row stride = lda
// elements). Lane (mrow, kh): two contiguous 8-element groups at
// k = kc + 8*kh and k = kc + 16 + 8*kh  (ISA 16-bit A layout).
__device__ __forceinline__ v16bf load_a_frag(const __bf16* base, int row,
                                             int lda, int kc, int mrow, int kh) {
  const __bf16* p = base + (size_t)(row + mrow) * lda + kc + 8 * kh;
  v8bf a0 = *(const v8bf*)p;
  v8bf a1 = *(const v8bf*)(p + 16);
  v16bf a;
#pragma unroll
  for (int i = 0; i < 8; ++i) { a[i] = a0[i]; a[8 + i] = a1[i]; }
  return a;
}

// B fragment: 32x16 bf16 tile. Weights are pre-transposed to [Ncols][K]
// row-major; a lane's k-values are two 8-element groups at
// k = kc + 8*kh and k = kc + 16 + 8*kh (same order as A).
__device__ __forceinline__ v16bf load_b_frag(const __bf16* wt, int ncol,
                                             int K, int kc, int kh) {
  const __bf16* p = wt + (size_t)ncol * K + kc + 8 * kh;
  v8bf b0 = *(const v8bf*)p;
  v8bf b1 = *(const v8bf*)(p + 16);
  v16bf b;
#pragma unroll
  for (int i = 0; i < 8; ++i) { b[i] = b0[i]; b[8 + i] = b1[i]; }
  return b;
}

// ---------------------------------------------------------------------------
// Utility kernels
// ---------------------------------------------------------------------------
__global__ void zero_f32_kernel(float* __restrict__ p, int n) {
  int i = blockIdx.x * blockDim.x + threadIdx.x;
  if (i < n) p[i] = 0.f;
}

// Transpose-convert f32 [K][Nc] row-major -> bf16 [Nc][K] row-major.
__global__ void convert_w_kernel(const float* __restrict__ src,
                                 __bf16* __restrict__ dst, int K, int Nc) {
  int i = blockIdx.x * blockDim.x + threadIdx.x;
  if (i < K * Nc) {
    int k = i / Nc, n = i - k * Nc;
    dst[(size_t)n * K + k] = (__bf16)src[i];
  }
}

// ---------------------------------------------------------------------------
// Edge kernel: 64 edges / block, 256 threads (8 wave32, 16 cols each,
// 4 row-tiles per wave).
//   m = cat(x[dst], x[src], ea)            [64 x 384] bf16 (LDS)
//   h = relu(m @ W1 + b1) @ W2 + b2        via WMMA, f32 accum
//   ea_new = ea + LN(h); scatter-add ea_new into agg[src]
// ---------------------------------------------------------------------------
__global__ __launch_bounds__(256) void edge_kernel(
    const float* __restrict__ xbuf, float* __restrict__ eabuf,
    const int* __restrict__ srcIdx, const int* __restrict__ dstIdx,
    const __bf16* __restrict__ W1t, const float* __restrict__ b1,
    const __bf16* __restrict__ W2t, const float* __restrict__ b2,
    const float* __restrict__ G, const float* __restrict__ Bt,
    float* __restrict__ agg) {
  constexpr int D = GNN_D, K1 = 3 * GNN_D, ROWS = 64, RT = ROWS / 16;
  // LDS layout (98816 B total):
  __shared__ __align__(16) unsigned char smem[98816];
  __bf16* mA  = (__bf16*)smem;                 // [64][384] bf16 (49152 B)
  float*  hF  = (float*)smem;                  // [64][128] f32, aliases mA
  float*  eaF = (float*)(smem + 49152);        // [64][128] f32 (32768 B)
  __bf16* h1  = (__bf16*)(smem + 81920);       // [64][128] bf16 (16384 B)
  int*    sL  = (int*)(smem + 98304);          // [64]
  int*    dL  = sL + 64;                       // [64]

  const int t  = threadIdx.x;
  const int e0 = blockIdx.x * ROWS;

  if (t < ROWS) {
    int e  = e0 + t;
    int ok = (e < GNN_E);
    sL[t] = ok ? srcIdx[e] : 0;
    dL[t] = ok ? dstIdx[e] : 0;
  }
  __syncthreads();

  // Gather cat(x_i, x_j, ea) -> bf16 LDS; keep ea in f32 for the residual.
#pragma unroll
  for (int task = t; task < ROWS * 32; task += 256) {
    int row = task >> 5;
    int c   = (task & 31) * 4;
    int e   = e0 + row;
    int ec  = e < GNN_E ? e : 0;
    int s = sL[row], d = dL[row];
    float4 xi = *(const float4*)(xbuf + (size_t)d * D + c);
    float4 xj = *(const float4*)(xbuf + (size_t)s * D + c);
    float4 ev = *(const float4*)(eabuf + (size_t)ec * D + c);
    *(v4bf*)(mA + (size_t)row * K1 + c) =
        (v4bf){(__bf16)xi.x, (__bf16)xi.y, (__bf16)xi.z, (__bf16)xi.w};
    *(v4bf*)(mA + (size_t)row * K1 + D + c) =
        (v4bf){(__bf16)xj.x, (__bf16)xj.y, (__bf16)xj.z, (__bf16)xj.w};
    *(v4bf*)(mA + (size_t)row * K1 + 2 * D + c) =
        (v4bf){(__bf16)ev.x, (__bf16)ev.y, (__bf16)ev.z, (__bf16)ev.w};
    *(float4*)(eaF + (size_t)row * D + c) = ev;
  }
  __syncthreads();

  const int lane = t & 31, wv = t >> 5;
  const int mrow = lane & 15, kh = lane >> 4;
  const int col  = wv * 16 + mrow;

  // GEMM1: [64 x 384] @ [384 x 128]
  {
    v8f acc[RT];
#pragma unroll
    for (int rt = 0; rt < RT; ++rt) acc[rt] = (v8f){};
#pragma unroll
    for (int kc = 0; kc < K1; kc += 32) {
      v16bf bf = load_b_frag(W1t, col, K1, kc, kh);
#pragma unroll
      for (int rt = 0; rt < RT; ++rt) {
        v16bf af = load_a_frag(mA, rt * 16, K1, kc, mrow, kh);
        acc[rt] = wmma_bf16(af, bf, acc[rt]);
      }
    }
    float bv = b1[col];
#pragma unroll
    for (int rt = 0; rt < RT; ++rt)
#pragma unroll
      for (int r = 0; r < 8; ++r) {
        int m = rt * 16 + r + 8 * kh;
        h1[(size_t)m * D + col] = (__bf16)fmaxf(acc[rt][r] + bv, 0.f);
      }
  }
  __syncthreads();   // h1 ready; mA dead -> hF may overlay it

  // GEMM2: [64 x 128] @ [128 x 128]
  {
    v8f acc[RT];
#pragma unroll
    for (int rt = 0; rt < RT; ++rt) acc[rt] = (v8f){};
#pragma unroll
    for (int kc = 0; kc < D; kc += 32) {
      v16bf bf = load_b_frag(W2t, col, D, kc, kh);
#pragma unroll
      for (int rt = 0; rt < RT; ++rt) {
        v16bf af = load_a_frag(h1, rt * 16, D, kc, mrow, kh);
        acc[rt] = wmma_bf16(af, bf, acc[rt]);
      }
    }
    float bv = b2[col];
#pragma unroll
    for (int rt = 0; rt < RT; ++rt)
#pragma unroll
      for (int r = 0; r < 8; ++r) {
        int m = rt * 16 + r + 8 * kh;
        hF[(size_t)m * D + col] = acc[rt][r] + bv;
      }
  }
  __syncthreads();

  // LayerNorm + residual + scatter-add. Each wave owns rows wv, wv+8, ...
  for (int row = wv; row < ROWS; row += 8) {
    float v[4], s = 0.f, s2 = 0.f;
#pragma unroll
    for (int j = 0; j < 4; ++j) {
      int c = lane + 32 * j;
      float h = hF[(size_t)row * D + c];
      v[j] = h; s += h; s2 += h * h;
    }
#pragma unroll
    for (int m = 16; m >= 1; m >>= 1) {
      s  += __shfl_xor(s, m, 32);
      s2 += __shfl_xor(s2, m, 32);
    }
    float mu = s * (1.f / D);
    float rs = rsqrtf(s2 * (1.f / D) - mu * mu + 1e-5f);
    int e = e0 + row;
    if (e < GNN_E) {
      int si = sL[row];
#pragma unroll
      for (int j = 0; j < 4; ++j) {
        int c = lane + 32 * j;
        float y = (v[j] - mu) * rs * G[c] + Bt[c];
        float o = eaF[(size_t)row * D + c] + y;
        eabuf[(size_t)e * D + c] = o;
        atomicAdd(agg + (size_t)si * D + c, o);
      }
    }
  }
}

// ---------------------------------------------------------------------------
// Node kernel: 64 nodes / block.  u = cat(x, agg)  [64 x 256]
//   x_new = x + LN(relu(u @ W1 + b1) @ W2 + b2)
// ---------------------------------------------------------------------------
__global__ __launch_bounds__(256) void node_kernel(
    float* __restrict__ xbuf, const float* __restrict__ agg,
    const __bf16* __restrict__ W1t, const float* __restrict__ b1,
    const __bf16* __restrict__ W2t, const float* __restrict__ b2,
    const float* __restrict__ G, const float* __restrict__ Bt) {
  constexpr int D = GNN_D, K1 = 2 * GNN_D, ROWS = 64, RT = ROWS / 16;
  // LDS layout (81920 B total):
  __shared__ __align__(16) unsigned char smem[81920];
  __bf16* uA = (__bf16*)smem;                  // [64][256] bf16 (32768 B)
  float*  hF = (float*)smem;                   // [64][128] f32, aliases uA
  float*  xF = (float*)(smem + 32768);         // [64][128] f32 (32768 B)
  __bf16* h1 = (__bf16*)(smem + 65536);        // [64][128] bf16 (16384 B)

  const int t  = threadIdx.x;
  const int n0 = blockIdx.x * ROWS;

  // Gather cat(x, agg) -> bf16 LDS; keep x in f32 for the residual.
#pragma unroll
  for (int task = t; task < ROWS * 64; task += 256) {
    int row = task >> 6;
    int c   = (task & 63) * 4;
    int n   = n0 + row;
    int nc  = n < GNN_N ? n : 0;
    if (c < D) {
      float4 xv = *(const float4*)(xbuf + (size_t)nc * D + c);
      *(v4bf*)(uA + (size_t)row * K1 + c) =
          (v4bf){(__bf16)xv.x, (__bf16)xv.y, (__bf16)xv.z, (__bf16)xv.w};
      *(float4*)(xF + (size_t)row * D + c) = xv;
    } else {
      float4 av = *(const float4*)(agg + (size_t)nc * D + (c - D));
      *(v4bf*)(uA + (size_t)row * K1 + c) =
          (v4bf){(__bf16)av.x, (__bf16)av.y, (__bf16)av.z, (__bf16)av.w};
    }
  }
  __syncthreads();

  const int lane = t & 31, wv = t >> 5;
  const int mrow = lane & 15, kh = lane >> 4;
  const int col  = wv * 16 + mrow;

  // GEMM1: [64 x 256] @ [256 x 128]
  {
    v8f acc[RT];
#pragma unroll
    for (int rt = 0; rt < RT; ++rt) acc[rt] = (v8f){};
#pragma unroll
    for (int kc = 0; kc < K1; kc += 32) {
      v16bf bf = load_b_frag(W1t, col, K1, kc, kh);
#pragma unroll
      for (int rt = 0; rt < RT; ++rt) {
        v16bf af = load_a_frag(uA, rt * 16, K1, kc, mrow, kh);
        acc[rt] = wmma_bf16(af, bf, acc[rt]);
      }
    }
    float bv = b1[col];
#pragma unroll
    for (int rt = 0; rt < RT; ++rt)
#pragma unroll
      for (int r = 0; r < 8; ++r) {
        int m = rt * 16 + r + 8 * kh;
        h1[(size_t)m * D + col] = (__bf16)fmaxf(acc[rt][r] + bv, 0.f);
      }
  }
  __syncthreads();   // h1 ready; uA dead -> hF may overlay it

  // GEMM2: [64 x 128] @ [128 x 128]
  {
    v8f acc[RT];
#pragma unroll
    for (int rt = 0; rt < RT; ++rt) acc[rt] = (v8f){};
#pragma unroll
    for (int kc = 0; kc < D; kc += 32) {
      v16bf bf = load_b_frag(W2t, col, D, kc, kh);
#pragma unroll
      for (int rt = 0; rt < RT; ++rt) {
        v16bf af = load_a_frag(h1, rt * 16, D, kc, mrow, kh);
        acc[rt] = wmma_bf16(af, bf, acc[rt]);
      }
    }
    float bv = b2[col];
#pragma unroll
    for (int rt = 0; rt < RT; ++rt)
#pragma unroll
      for (int r = 0; r < 8; ++r) {
        int m = rt * 16 + r + 8 * kh;
        hF[(size_t)m * D + col] = acc[rt][r] + bv;
      }
  }
  __syncthreads();

  // LayerNorm + residual.
  for (int row = wv; row < ROWS; row += 8) {
    float v[4], s = 0.f, s2 = 0.f;
#pragma unroll
    for (int j = 0; j < 4; ++j) {
      int c = lane + 32 * j;
      float h = hF[(size_t)row * D + c];
      v[j] = h; s += h; s2 += h * h;
    }
#pragma unroll
    for (int m = 16; m >= 1; m >>= 1) {
      s  += __shfl_xor(s, m, 32);
      s2 += __shfl_xor(s2, m, 32);
    }
    float mu = s * (1.f / D);
    float rs = rsqrtf(s2 * (1.f / D) - mu * mu + 1e-5f);
    int n = n0 + row;
    if (n < GNN_N) {
#pragma unroll
      for (int j = 0; j < 4; ++j) {
        int c = lane + 32 * j;
        float y = (v[j] - mu) * rs * G[c] + Bt[c];
        xbuf[(size_t)n * D + c] = xF[(size_t)row * D + c] + y;
      }
    }
  }
}

// ---------------------------------------------------------------------------
// Host launcher
// ---------------------------------------------------------------------------
extern "C" void kernel_launch(void* const* d_in, const int* in_sizes, int n_in,
                              void* d_out, int out_size, void* d_ws, size_t ws_size,
                              hipStream_t stream) {
  (void)in_sizes; (void)n_in; (void)out_size; (void)ws_size;
  constexpr int L = GNN_L, N = GNN_N, E = GNN_E, D = GNN_D;

  const float* x_in   = (const float*)d_in[0];
  const int*   eidx   = (const int*)d_in[1];
  const float* ea_in  = (const float*)d_in[2];
  const float* eW1    = (const float*)d_in[3];
  const float* eb1    = (const float*)d_in[4];
  const float* eW2    = (const float*)d_in[5];
  const float* eb2    = (const float*)d_in[6];
  const float* eG     = (const float*)d_in[7];
  const float* eB     = (const float*)d_in[8];
  const float* nW1    = (const float*)d_in[9];
  const float* nb1    = (const float*)d_in[10];
  const float* nW2    = (const float*)d_in[11];
  const float* nb2    = (const float*)d_in[12];
  const float* nG     = (const float*)d_in[13];
  const float* nB     = (const float*)d_in[14];

  const int* srcIdx = eidx;       // edge_index[0]: x_j source + scatter target
  const int* dstIdx = eidx + E;   // edge_index[1]: x_i

  float* xbuf  = (float*)d_out;                 // [N][D] f32 (output 0)
  float* eabuf = xbuf + (size_t)N * D;          // [E][D] f32 (output 1)

  float*  agg   = (float*)d_ws;                               // [N][D] f32
  __bf16* wbase = (__bf16*)((char*)d_ws + (size_t)N * D * 4);
  __bf16* eW1t  = wbase;                         // L x [128][384]
  __bf16* eW2t  = eW1t + (size_t)L * D * 3 * D;  // L x [128][128]
  __bf16* nW1t  = eW2t + (size_t)L * D * D;      // L x [128][256]
  __bf16* nW2t  = nW1t + (size_t)L * D * 2 * D;  // L x [128][128]

  // Seed outputs with the inputs (layers update in place).
  hipMemcpyAsync(xbuf, x_in, (size_t)N * D * sizeof(float),
                 hipMemcpyDeviceToDevice, stream);
  hipMemcpyAsync(eabuf, ea_in, (size_t)E * D * sizeof(float),
                 hipMemcpyDeviceToDevice, stream);

  // Convert+transpose weights to bf16 [Ncols][K].
  for (int l = 0; l < L; ++l) {
    int n1 = 3 * D * D, n2 = D * D, n3 = 2 * D * D;
    convert_w_kernel<<<(n1 + 255) / 256, 256, 0, stream>>>(
        eW1 + (size_t)l * n1, eW1t + (size_t)l * n1, 3 * D, D);
    convert_w_kernel<<<(n2 + 255) / 256, 256, 0, stream>>>(
        eW2 + (size_t)l * n2, eW2t + (size_t)l * n2, D, D);
    convert_w_kernel<<<(n3 + 255) / 256, 256, 0, stream>>>(
        nW1 + (size_t)l * n3, nW1t + (size_t)l * n3, 2 * D, D);
    convert_w_kernel<<<(n2 + 255) / 256, 256, 0, stream>>>(
        nW2 + (size_t)l * n2, nW2t + (size_t)l * n2, D, D);
  }

  for (int l = 0; l < L; ++l) {
    zero_f32_kernel<<<(N * D + 255) / 256, 256, 0, stream>>>(agg, N * D);
    edge_kernel<<<(E + 63) / 64, 256, 0, stream>>>(
        xbuf, eabuf, srcIdx, dstIdx,
        eW1t + (size_t)l * 3 * D * D, eb1 + (size_t)l * D,
        eW2t + (size_t)l * D * D,     eb2 + (size_t)l * D,
        eG + (size_t)l * D,           eB + (size_t)l * D, agg);
    node_kernel<<<(N + 63) / 64, 256, 0, stream>>>(
        xbuf, agg,
        nW1t + (size_t)l * 2 * D * D, nb1 + (size_t)l * D,
        nW2t + (size_t)l * D * D,     nb2 + (size_t)l * D,
        nG + (size_t)l * D,           nB + (size_t)l * D);
  }
}
